// HLmEncoderLayer_40346922779145
// MI455X (gfx1250) — compile-verified, matrix operand order checked by reading, not run on hardware
//
#include <hip/hip_runtime.h>
#include <hip/hip_bf16.h>
#include <math.h>

// ---------------------------------------------------------------------------
// MI455X (gfx1250) implementation: bf16 WMMA (v_wmma_f32_16x16x32_bf16) for
// all GEMMs + flash attention; fp32 accumulation throughout.  GEMM LDS tiles
// are staged with the CDNA5 Tensor Data Mover (tensor_load_to_lds, TENSORcnt)
// when the builtin is available, double-buffered to overlap DMA with WMMA.
// ---------------------------------------------------------------------------

typedef __bf16 bf16x16 __attribute__((ext_vector_type(16)));
typedef float  f32x8   __attribute__((ext_vector_type(8)));
typedef unsigned int u32x4 __attribute__((ext_vector_type(4)));
typedef int          i32x4 __attribute__((ext_vector_type(4)));
typedef int          i32x8 __attribute__((ext_vector_type(8)));

union FragBf { uint4 u[2]; bf16x16 v; };

#if __has_builtin(__builtin_amdgcn_tensor_load_to_lds) && \
    __has_builtin(__builtin_amdgcn_s_wait_tensorcnt)
#define USE_TDM 1
#else
#define USE_TDM 0
#endif

// ---- problem constants ----
constexpr int B_    = 2;
constexpr int S_    = 2048;
constexpr int D_    = 1024;
constexpr int NH_   = 16;
constexpr int DH_   = 64;
constexpr int NIAF_ = 4;
constexpr int QKV_  = 1024;
constexpr int MLP_  = 4096;
constexpr int Z_    = 32;
constexpr int HIAF_ = 128;
constexpr int R_    = 8;
constexpr int M_    = B_ * S_;                    // 4096 tokens
constexpr int UPN_  = QKV_ + 2 * QKV_ + 2 * MLP_; // 11264
constexpr int XZK_  = D_ + Z_;                    // 1056
constexpr int DWK_  = Z_ + QKV_ + MLP_;           // 5152
constexpr int TK_   = R_ + S_;                    // 2056 real keys
constexpr int TKP_  = 2112;                       // padded keys (66 tiles of 32)
constexpr float EPS_     = 1e-5f;
constexpr float ZSCALE_  = 0.04419417382415922f;  // sqrt(1/(Z*NL)) = sqrt(1/512)
constexpr float SPSHIFT_ = 0.5413248546129181f;   // log(e-1)
constexpr float NEGINF_  = -1e9f;

static __device__ __forceinline__ unsigned short f2bf(float f) {
  union { float f; unsigned int u; } v; v.f = f;
  unsigned int r = v.u + 0x7FFFu + ((v.u >> 16) & 1u);   // round-nearest-even
  return (unsigned short)(r >> 16);
}

static __device__ __forceinline__ f32x8 wmma_bf16(const FragBf& a, const FragBf& b, f32x8 c) {
  return __builtin_amdgcn_wmma_f32_16x16x32_bf16(false, a.v, false, b.v, (short)0, c, false, false);
}

static __device__ __forceinline__ float dot32(const float* nn, const float* __restrict__ w) {
  float a = 0.f;
#pragma unroll
  for (int c = 0; c < 32; ++c) a += nn[c] * w[c];
  return a;
}

#if USE_TDM
// Issue one TDM descriptor: load a 128-row x 32-col bf16 tile (row stride
// `strideElems` in memory) into LDS at byte address `lds_addr`, padding the
// LDS destination by 16B after each 64B row (-> 40-short LDS row pitch).
// D# bitfields per CDNA5 ISA ch.8: group0 {count,lds,global,type}, group1
// {data_size=2B, pad 4 DW every 16 DW, tensor 32x128, tile 32x128, stride}.
static __device__ __forceinline__ void tdm_load_tile(unsigned lds_addr,
                                                     const unsigned short* gsrc,
                                                     int strideElems) {
  const unsigned long long ga = (unsigned long long)(uintptr_t)gsrc;
  u32x4 g0;
  g0[0] = 1u;                                              // count=1, user mode
  g0[1] = lds_addr;                                        // lds_addr [63:32]
  g0[2] = (unsigned)(ga & 0xffffffffu);                    // global_addr lo
  g0[3] = (unsigned)((ga >> 32) & 0x01ffffffu) | (2u << 30); // addr hi | type=2
  i32x8 g1;
  g1[0] = (1 << 16) | (1 << 20) | (3 << 22) | (3 << 25);   // 2B, pad_en, 16DW/4DW
  g1[1] = (int)(32u << 16);                                // tensor_dim0 = 32
  g1[2] = (int)(128u << 16);                               // tensor_dim1 = 128
  g1[3] = (int)(32u << 16);                                // tile_dim0 = 32
  g1[4] = 128;                                             // tile_dim1 = 128
  g1[5] = strideElems;                                     // tensor_dim0_stride
  g1[6] = 0;
  g1[7] = 0;
  i32x4 gz4;
  gz4[0] = 0; gz4[1] = 0; gz4[2] = 0; gz4[3] = 0;
  i32x8 gz8;
  gz8[0] = 0; gz8[1] = 0; gz8[2] = 0; gz8[3] = 0;
  gz8[4] = 0; gz8[5] = 0; gz8[6] = 0; gz8[7] = 0;
  // this toolchain's builtin takes 6 args (g0, g1, g2, g3, extra, cpol)
  __builtin_amdgcn_tensor_load_to_lds(g0, g1, gz4, gz4, gz8, 0);
}
#endif

// ---------------------------------------------------------------------------
// fp32 -> bf16 conversion (grid-stride)
// ---------------------------------------------------------------------------
__global__ void f32_to_bf16_kernel(const float* __restrict__ src,
                                   unsigned short* __restrict__ dst, size_t n) {
  size_t i = (size_t)blockIdx.x * blockDim.x + threadIdx.x;
  size_t stride = (size_t)gridDim.x * blockDim.x;
  for (; i < n; i += stride) dst[i] = f2bf(src[i]);
}

// ---------------------------------------------------------------------------
// LayerNorm + IAF posterior. One block per token.
// ---------------------------------------------------------------------------
__global__ __launch_bounds__(256) void ln_iaf_kernel(
    const float* __restrict__ h, const unsigned char* __restrict__ mask,
    const float* __restrict__ noise,
    const float* __restrict__ ln_w, const float* __restrict__ ln_b,
    const float* __restrict__ iaf_wx, const float* __restrict__ iaf_wn,
    const float* __restrict__ iaf_w2,
    unsigned short* __restrict__ xz_bf, unsigned short* __restrict__ down_in,
    float* __restrict__ out_z, float* __restrict__ out_mu,
    float* __restrict__ out_sigma) {
  const int m = blockIdx.x;
  const int tid = threadIdx.x;
  __shared__ float xs[D_];
  __shared__ float red[256];
  __shared__ float nz[Z_];
  __shared__ float hdn[HIAF_];

  const float* hr = h + (size_t)m * D_;
  float v4[4];
  float s1 = 0.f;
#pragma unroll
  for (int k = 0; k < 4; ++k) { v4[k] = hr[tid + 256 * k]; s1 += v4[k]; }
  red[tid] = s1; __syncthreads();
  for (int off = 128; off > 0; off >>= 1) {
    if (tid < off) red[tid] += red[tid + off];
    __syncthreads();
  }
  const float mean = red[0] * (1.0f / D_);
  __syncthreads();
  float s2 = 0.f;
#pragma unroll
  for (int k = 0; k < 4; ++k) { float d = v4[k] - mean; s2 += d * d; }
  red[tid] = s2; __syncthreads();
  for (int off = 128; off > 0; off >>= 1) {
    if (tid < off) red[tid] += red[tid + off];
    __syncthreads();
  }
  const float rstd = rsqrtf(red[0] * (1.0f / D_) + EPS_);
  const float fm = mask[m] ? 1.0f : 0.0f;
#pragma unroll
  for (int k = 0; k < 4; ++k) {
    int d = tid + 256 * k;
    float x = (v4[k] - mean) * rstd * ln_w[d] + ln_b[d];
    xs[d] = x;
    xz_bf[(size_t)m * XZK_ + d] = f2bf(x);
  }
  if (tid < Z_) nz[tid] = noise[(size_t)m * Z_ + tid] * fm;
  __syncthreads();

  if (tid < HIAF_) {
    float acc = 0.f;
    const float* wr = iaf_wx + (size_t)tid * D_;
    for (int d = 0; d < D_; ++d) acc += xs[d] * wr[d];
    const int degh = tid % (Z_ - 1);                 // deg_h - 1
    const float* wn = iaf_wn + (size_t)tid * Z_;
    for (int c = 0; c <= degh; ++c) acc += nz[c] * wn[c];  // M1: deg_h >= deg_in
    hdn[tid] = acc / (1.0f + __expf(-acc));          // silu
  }
  __syncthreads();

  if (tid < Z_) {
    float amu = 0.f, als = 0.f;
    const float* wmu = iaf_w2 + (size_t)tid * HIAF_;
    const float* wls = iaf_w2 + (size_t)(tid + Z_) * HIAF_;
    for (int j = 0; j < HIAF_; ++j) {
      if ((j % (Z_ - 1)) < tid) {                    // M2: deg_out > deg_h
        amu += hdn[j] * wmu[j];
        als += hdn[j] * wls[j];
      }
    }
    const float zs = ZSCALE_ * fm;
    amu *= zs; als *= zs;
    const float spi = als + SPSHIFT_;
    const float sigma = (spi > 20.f) ? spi : log1pf(__expf(spi));
    const float z = amu + sigma * nz[tid];
    const size_t zi = (size_t)m * Z_ + tid;
    out_mu[zi] = amu;
    out_sigma[zi] = sigma;
    out_z[zi] = z;
    const unsigned short zb = f2bf(z);
    xz_bf[(size_t)m * XZK_ + D_ + tid] = zb;
    down_in[(size_t)m * DWK_ + tid] = zb;
  }
}

// ---------------------------------------------------------------------------
// Tiled bf16 WMMA GEMM: C[M,N] = A[M,K] * W[N,K]^T  (fp32 accumulate).
// Block tile 128x128, K-step 32, 8 waves, each wave 32(M) x 64(N) = 2x4 WMMA.
// LDS staging: TDM double-buffered (wave 0 issues the DMA) when available,
// else register-double-buffered vector loads.
// mode 0: store fp32 C.  mode 1: Out = Hres + alpha * C  (fused residual).
// ---------------------------------------------------------------------------
__global__ __launch_bounds__(256) void gemm_bf16_kernel(
    const unsigned short* __restrict__ A, const unsigned short* __restrict__ W,
    float* __restrict__ Cout, const float* __restrict__ Hres,
    const float* __restrict__ alphap, float* __restrict__ Out,
    int Mdim, int Ndim, int Kdim, int mode) {
#if USE_TDM
  __shared__ unsigned short As[2][128 * 40];
  __shared__ unsigned short Ws[2][128 * 40];
#else
  __shared__ unsigned short As[1][128 * 40];
  __shared__ unsigned short Ws[1][128 * 40];
#endif
  const int tid = threadIdx.x;
  const int lane = tid & 31;
  const int wv = tid >> 5;
  const int wm = wv >> 1;          // 0..3 -> 32 M-rows each
  const int wn = wv & 1;           // 0..1 -> 64 N-cols each
  const int h2 = lane >> 4;
  const int l16 = lane & 15;
  const int m0 = blockIdx.y * 128;
  const int n0 = blockIdx.x * 128;
  const int nk = Kdim >> 5;

  f32x8 acc[2][4];
#pragma unroll
  for (int mi = 0; mi < 2; ++mi)
#pragma unroll
    for (int ni = 0; ni < 4; ++ni)
#pragma unroll
      for (int r = 0; r < 8; ++r) acc[mi][ni][r] = 0.f;

#if USE_TDM
  const unsigned short* Agp = A + (size_t)m0 * Kdim;
  const unsigned short* Wgp = W + (size_t)n0 * Kdim;
  if (wv == 0) {
    tdm_load_tile((unsigned)(uintptr_t)&As[0][0], Agp, Kdim);
    tdm_load_tile((unsigned)(uintptr_t)&Ws[0][0], Wgp, Kdim);
  }
  for (int i = 0; i < nk; ++i) {
    __syncthreads();                       // everyone done reading buf (i+1)&1
    if (wv == 0) {
      if (i + 1 < nk) {
        const int kn = (i + 1) << 5;
        tdm_load_tile((unsigned)(uintptr_t)&As[(i + 1) & 1][0], Agp + kn, Kdim);
        tdm_load_tile((unsigned)(uintptr_t)&Ws[(i + 1) & 1][0], Wgp + kn, Kdim);
        __builtin_amdgcn_s_wait_tensorcnt((short)2);  // tile i complete
      } else {
        __builtin_amdgcn_s_wait_tensorcnt((short)0);
      }
    }
    __syncthreads();                       // tile i visible to all waves
    const unsigned short* Ab = As[i & 1];
    const unsigned short* Wb = Ws[i & 1];
    FragBf a[2], bw[4];
#pragma unroll
    for (int mi = 0; mi < 2; ++mi) {
      const unsigned short* base = Ab + (wm * 32 + mi * 16 + l16) * 40;
      a[mi].u[0] = *(const uint4*)(base + 8 * h2);
      a[mi].u[1] = *(const uint4*)(base + 16 + 8 * h2);
    }
#pragma unroll
    for (int ni = 0; ni < 4; ++ni) {
      const unsigned short* base = Wb + (wn * 64 + ni * 16 + l16) * 40;
      bw[ni].u[0] = *(const uint4*)(base + 8 * h2);
      bw[ni].u[1] = *(const uint4*)(base + 16 + 8 * h2);
    }
#pragma unroll
    for (int mi = 0; mi < 2; ++mi)
#pragma unroll
      for (int ni = 0; ni < 4; ++ni)
        acc[mi][ni] = wmma_bf16(a[mi], bw[ni], acc[mi][ni]);
  }
#else
  // register-double-buffered fallback: each thread owns 2 16B chunks of the
  // A tile and 2 of the W tile (128x32 bf16 = 512 chunks of 8 elements).
  const int r0 = tid >> 2,         c0 = (tid & 3) << 3;
  const int r1 = (tid + 256) >> 2, c1 = ((tid + 256) & 3) << 3;
  uint4 pa0 = *(const uint4*)(A + (size_t)(m0 + r0) * Kdim + c0);
  uint4 pa1 = *(const uint4*)(A + (size_t)(m0 + r1) * Kdim + c1);
  uint4 pw0 = *(const uint4*)(W + (size_t)(n0 + r0) * Kdim + c0);
  uint4 pw1 = *(const uint4*)(W + (size_t)(n0 + r1) * Kdim + c1);
  for (int i = 0; i < nk; ++i) {
    *(uint4*)(As[0] + r0 * 40 + c0) = pa0;
    *(uint4*)(As[0] + r1 * 40 + c1) = pa1;
    *(uint4*)(Ws[0] + r0 * 40 + c0) = pw0;
    *(uint4*)(Ws[0] + r1 * 40 + c1) = pw1;
    __syncthreads();
    const bool more = (i + 1 < nk);
    uint4 na0{}, na1{}, nw0{}, nw1{};
    if (more) {
      const int kn = (i + 1) << 5;
      na0 = *(const uint4*)(A + (size_t)(m0 + r0) * Kdim + kn + c0);
      na1 = *(const uint4*)(A + (size_t)(m0 + r1) * Kdim + kn + c1);
      nw0 = *(const uint4*)(W + (size_t)(n0 + r0) * Kdim + kn + c0);
      nw1 = *(const uint4*)(W + (size_t)(n0 + r1) * Kdim + kn + c1);
    }
    FragBf a[2], bw[4];
#pragma unroll
    for (int mi = 0; mi < 2; ++mi) {
      const unsigned short* base = As[0] + (wm * 32 + mi * 16 + l16) * 40;
      a[mi].u[0] = *(const uint4*)(base + 8 * h2);
      a[mi].u[1] = *(const uint4*)(base + 16 + 8 * h2);
    }
#pragma unroll
    for (int ni = 0; ni < 4; ++ni) {
      const unsigned short* base = Ws[0] + (wn * 64 + ni * 16 + l16) * 40;
      bw[ni].u[0] = *(const uint4*)(base + 8 * h2);
      bw[ni].u[1] = *(const uint4*)(base + 16 + 8 * h2);
    }
#pragma unroll
    for (int mi = 0; mi < 2; ++mi)
#pragma unroll
      for (int ni = 0; ni < 4; ++ni)
        acc[mi][ni] = wmma_bf16(a[mi], bw[ni], acc[mi][ni]);
    __syncthreads();
    pa0 = na0; pa1 = na1; pw0 = nw0; pw1 = nw1;
  }
#endif

  const float alpha = (mode == 1) ? alphap[0] : 0.f;
#pragma unroll
  for (int mi = 0; mi < 2; ++mi)
#pragma unroll
    for (int ni = 0; ni < 4; ++ni)
#pragma unroll
      for (int r = 0; r < 8; ++r) {
        const int mrow = m0 + wm * 32 + mi * 16 + r + 8 * h2;
        const int ncol = n0 + wn * 64 + ni * 16 + l16;
        const size_t idx = (size_t)mrow * Ndim + ncol;
        const float val = acc[mi][ni][r];
        if (mode == 0) Cout[idx] = val;
        else           Out[idx]  = Hres[idx] + alpha * val;
      }
}

// ---------------------------------------------------------------------------
// RoPE + IAF-up noise injection + pack q/k/v. One block per token.
//   Qb: [B*NH][S][64]   Kb: [B*NH][TKP][64]   Vt: [B*NH][64][TKP] (transposed)
// ---------------------------------------------------------------------------
__global__ __launch_bounds__(256) void rope_pack_kernel(
    const float* __restrict__ u, const float* __restrict__ next_noise,
    const float* __restrict__ iaf_up_w,
    unsigned short* __restrict__ Qb, unsigned short* __restrict__ Kb,
    unsigned short* __restrict__ Vt) {
  const int m = blockIdx.x;
  const int b = m / S_;
  const int s = m % S_;
  const int tid = threadIdx.x;
  __shared__ float nn[Z_];
  if (tid < Z_) nn[tid] = next_noise[(size_t)m * Z_ + tid];
  __syncthreads();

  const float* ur = u + (size_t)m * UPN_;
  const float linv = -logf(10000.0f) * (2.0f / DH_);

  for (int p = tid; p < 512; p += 256) {
    const int hh = p >> 5;
    const int dj = p & 31;
    float sn, cs;
    __sincosf((float)s * __expf(linv * (float)dj), &sn, &cs);
    {
      const float x1 = ur[hh * 64 + dj];
      const float x2 = ur[hh * 64 + dj + 32];
      unsigned short* q = Qb + ((size_t)(b * NH_ + hh) * S_ + s) * 64;
      q[dj]      = f2bf(x1 * cs - x2 * sn);
      q[dj + 32] = f2bf(x2 * cs + x1 * sn);
    }
    {
      float k1 = ur[QKV_ + hh * 64 + dj];
      float k2 = ur[QKV_ + hh * 64 + dj + 32];
      if (hh < NIAF_) {
        k1 += dot32(nn, iaf_up_w + (size_t)(hh * 64 + dj) * Z_);
        k2 += dot32(nn, iaf_up_w + (size_t)(hh * 64 + dj + 32) * Z_);
      }
      unsigned short* k = Kb + ((size_t)(b * NH_ + hh) * TKP_ + R_ + s) * 64;
      k[dj]      = f2bf(k1 * cs - k2 * sn);
      k[dj + 32] = f2bf(k2 * cs + k1 * sn);
    }
  }
  for (int p = tid; p < QKV_; p += 256) {
    const int hh = p >> 6;
    const int d = p & 63;
    float v = ur[2 * QKV_ + p];
    if (p < NIAF_ * DH_) v += dot32(nn, iaf_up_w + (size_t)(QKV_ + p) * Z_);
    Vt[((size_t)(b * NH_ + hh) * 64 + d) * TKP_ + R_ + s] = f2bf(v);
  }
}

// register tokens (t < R) and zero padding (t >= R+S) for K/V
__global__ __launch_bounds__(256) void fill_reg_pad_kernel(
    const float* __restrict__ reg_k, const float* __restrict__ reg_v,
    unsigned short* __restrict__ Kb, unsigned short* __restrict__ Vt) {
  const int bh = blockIdx.x;
  const int h = bh % NH_;
  const int tid = threadIdx.x;
  for (int idx = tid; idx < R_ * 64; idx += 256) {
    const int t = idx >> 6, d = idx & 63;
    Kb[((size_t)bh * TKP_ + t) * 64 + d] = f2bf(reg_k[((size_t)t * NH_ + h) * 64 + d]);
    Vt[((size_t)bh * 64 + d) * TKP_ + t] = f2bf(reg_v[((size_t)t * NH_ + h) * 64 + d]);
  }
  constexpr int PAD = TKP_ - TK_;   // 56
  for (int idx = tid; idx < PAD * 64; idx += 256) {
    const int t = TK_ + (idx >> 6), d = idx & 63;
    Kb[((size_t)bh * TKP_ + t) * 64 + d] = 0;
    Vt[((size_t)bh * 64 + d) * TKP_ + t] = 0;
  }
}

// ---------------------------------------------------------------------------
// SiLU-gated MLP: down_in[:, 1056:5152] = silu(gate) * val
// ---------------------------------------------------------------------------
__global__ __launch_bounds__(256) void mlp_kernel(
    const float* __restrict__ u, unsigned short* __restrict__ down_in) {
  const int m = blockIdx.x;
  const float* ur = u + (size_t)m * UPN_;
  for (int j = threadIdx.x; j < MLP_; j += 256) {
    const float g = ur[3 * QKV_ + j];
    const float v = ur[3 * QKV_ + MLP_ + j];
    down_in[(size_t)m * DWK_ + Z_ + QKV_ + j] = f2bf(g / (1.0f + __expf(-g)) * v);
  }
}

// ---------------------------------------------------------------------------
// Flash attention. grid = (B*NH, S/128); 8 waves, 16 query rows per wave.
// Streaming softmax over 32-key tiles; QK^T and P*V via WMMA; P's C->A
// re-layout goes through a per-wave LDS tile (uniform barriers).
// ---------------------------------------------------------------------------
__global__ __launch_bounds__(256) void attn_kernel(
    const unsigned short* __restrict__ Qb, const unsigned short* __restrict__ Kb,
    const unsigned short* __restrict__ Vt, const float* __restrict__ amask,
    unsigned short* __restrict__ down_in) {
  __shared__ unsigned short Ps[8 * 16 * 40];
  const int bh = blockIdx.x;
  const int b = bh / NH_;
  const int h = bh % NH_;
  const int tid = threadIdx.x;
  const int wv = tid >> 5;
  const int lane = tid & 31;
  const int h2 = lane >> 4;
  const int l16 = lane & 15;
  const int q0 = blockIdx.y * 128 + wv * 16;

  const unsigned short* Qp = Qb + (size_t)bh * S_ * 64;
  const unsigned short* Kp = Kb + (size_t)bh * TKP_ * 64;
  const unsigned short* Vp = Vt + (size_t)bh * 64 * TKP_;

  FragBf qa[2];
  {
    const unsigned short* row = Qp + (size_t)(q0 + l16) * 64;
    qa[0].u[0] = *(const uint4*)(row + 8 * h2);
    qa[0].u[1] = *(const uint4*)(row + 16 + 8 * h2);
    qa[1].u[0] = *(const uint4*)(row + 32 + 8 * h2);
    qa[1].u[1] = *(const uint4*)(row + 48 + 8 * h2);
  }

  float mrow[8], lrow[8];
  f32x8 o[4];
#pragma unroll
  for (int r = 0; r < 8; ++r) { mrow[r] = -3e38f; lrow[r] = 0.f; }
#pragma unroll
  for (int f = 0; f < 4; ++f)
#pragma unroll
    for (int r = 0; r < 8; ++r) o[f][r] = 0.f;

  unsigned short* myP = Ps + wv * 16 * 40;

  for (int t0 = 0; t0 < TKP_; t0 += 32) {
    f32x8 sc[2];
#pragma unroll
    for (int sub = 0; sub < 2; ++sub) {
#pragma unroll
      for (int r = 0; r < 8; ++r) sc[sub][r] = 0.f;
      const unsigned short* krow = Kp + (size_t)(t0 + sub * 16 + l16) * 64;
      FragBf kb0, kb1;
      kb0.u[0] = *(const uint4*)(krow + 8 * h2);
      kb0.u[1] = *(const uint4*)(krow + 16 + 8 * h2);
      kb1.u[0] = *(const uint4*)(krow + 32 + 8 * h2);
      kb1.u[1] = *(const uint4*)(krow + 48 + 8 * h2);
      sc[sub] = wmma_bf16(qa[0], kb0, sc[sub]);
      sc[sub] = wmma_bf16(qa[1], kb1, sc[sub]);
    }
#pragma unroll
    for (int sub = 0; sub < 2; ++sub) {
      const int t = t0 + sub * 16 + l16;
#pragma unroll
      for (int r = 0; r < 8; ++r) {
        const int qi = q0 + r + 8 * h2;
        float v = sc[sub][r] * 0.125f;     // 1/sqrt(DH)
        if (t >= R_) {
          const int j = t - R_;
          if (j < S_) {
            v += amask[((size_t)b * S_ + qi) * S_ + j];
            if (h < NIAF_ && j >= qi) v += NEGINF_;
          } else {
            v += NEGINF_;                  // padded keys
          }
        }
        sc[sub][r] = v;
      }
    }
    float mnew[8];
#pragma unroll
    for (int r = 0; r < 8; ++r) {
      float p = fmaxf(sc[0][r], sc[1][r]);
#pragma unroll
      for (int off = 8; off > 0; off >>= 1) p = fmaxf(p, __shfl_xor(p, off, 32));
      mnew[r] = fmaxf(mrow[r], p);
    }
#pragma unroll
    for (int r = 0; r < 8; ++r) {
      const float scale = __expf(mrow[r] - mnew[r]);
      lrow[r] *= scale;
#pragma unroll
      for (int f = 0; f < 4; ++f) o[f][r] *= scale;
      mrow[r] = mnew[r];
    }
#pragma unroll
    for (int sub = 0; sub < 2; ++sub)
#pragma unroll
      for (int r = 0; r < 8; ++r) sc[sub][r] = __expf(sc[sub][r] - mrow[r]);
#pragma unroll
    for (int r = 0; r < 8; ++r) {
      float p = sc[0][r] + sc[1][r];
#pragma unroll
      for (int off = 8; off > 0; off >>= 1) p += __shfl_xor(p, off, 32);
      lrow[r] += p;
    }
#pragma unroll
    for (int sub = 0; sub < 2; ++sub)
#pragma unroll
      for (int r = 0; r < 8; ++r)
        myP[(r + 8 * h2) * 40 + sub * 16 + l16] = f2bf(sc[sub][r]);
    __syncthreads();
    FragBf pa;
    {
      const unsigned short* prow = myP + l16 * 40;
      pa.u[0] = *(const uint4*)(prow + 8 * h2);
      pa.u[1] = *(const uint4*)(prow + 16 + 8 * h2);
    }
#pragma unroll
    for (int f = 0; f < 4; ++f) {
      const unsigned short* vrow = Vp + (size_t)(f * 16 + l16) * TKP_ + t0;
      FragBf vb;
      vb.u[0] = *(const uint4*)(vrow + 8 * h2);
      vb.u[1] = *(const uint4*)(vrow + 16 + 8 * h2);
      o[f] = wmma_bf16(pa, vb, o[f]);
    }
    __syncthreads();
  }

#pragma unroll
  for (int f = 0; f < 4; ++f)
#pragma unroll
    for (int r = 0; r < 8; ++r) {
      const int qi = q0 + r + 8 * h2;
      const size_t mtok = (size_t)b * S_ + qi;
      down_in[mtok * DWK_ + Z_ + h * 64 + f * 16 + l16] = f2bf(o[f][r] / lrow[r]);
    }
}

// ---------------------------------------------------------------------------
// host side
// ---------------------------------------------------------------------------
extern "C" void kernel_launch(void* const* d_in, const int* in_sizes, int n_in,
                              void* d_out, int out_size, void* d_ws, size_t ws_size,
                              hipStream_t stream) {
  (void)in_sizes; (void)n_in; (void)out_size; (void)ws_size;
  const float* h          = (const float*)d_in[0];
  const unsigned char* mk = (const unsigned char*)d_in[1];
  const float* noise      = (const float*)d_in[2];
  const float* next_noise = (const float*)d_in[3];
  const float* attn_mask  = (const float*)d_in[4];
  const float* ln_w       = (const float*)d_in[5];
  const float* ln_b       = (const float*)d_in[6];
  const float* alpha      = (const float*)d_in[7];
  const float* iaf_wx     = (const float*)d_in[8];
  const float* iaf_wn     = (const float*)d_in[9];
  const float* iaf_w2     = (const float*)d_in[10];
  const float* up_w       = (const float*)d_in[11];
  const float* iaf_up_w   = (const float*)d_in[12];
  const float* down_w     = (const float*)d_in[13];
  const float* reg_k      = (const float*)d_in[14];
  const float* reg_v      = (const float*)d_in[15];

  float* out      = (float*)d_out;
  float* out_h    = out;                               // [M, D]
  float* out_z    = out + (size_t)M_ * D_;             // [M, Z]
  float* out_mu   = out_z + (size_t)M_ * Z_;
  float* out_sig  = out_mu + (size_t)M_ * Z_;

  char* ws = (char*)d_ws;
  size_t off = 0;
  auto take = [&](size_t bytes) { char* p = ws + off; off += bytes; return p; };
  unsigned short* xz_bf   = (unsigned short*)take((size_t)M_ * XZK_ * 2);
  unsigned short* upw_bf  = (unsigned short*)take((size_t)UPN_ * XZK_ * 2);
  unsigned short* dww_bf  = (unsigned short*)take((size_t)D_ * DWK_ * 2);
  float*          u_buf   = (float*)take((size_t)M_ * UPN_ * 4);
  unsigned short* Qb      = (unsigned short*)take((size_t)B_ * NH_ * S_ * 64 * 2);
  unsigned short* Kb      = (unsigned short*)take((size_t)B_ * NH_ * TKP_ * 64 * 2);
  unsigned short* Vt      = (unsigned short*)take((size_t)B_ * NH_ * 64 * TKP_ * 2);
  unsigned short* dn_bf   = (unsigned short*)take((size_t)M_ * DWK_ * 2);

  f32_to_bf16_kernel<<<4096, 256, 0, stream>>>(up_w, upw_bf, (size_t)UPN_ * XZK_);
  f32_to_bf16_kernel<<<4096, 256, 0, stream>>>(down_w, dww_bf, (size_t)D_ * DWK_);

  ln_iaf_kernel<<<M_, 256, 0, stream>>>(h, mk, noise, ln_w, ln_b, iaf_wx, iaf_wn,
                                        iaf_w2, xz_bf, dn_bf, out_z, out_mu, out_sig);

  gemm_bf16_kernel<<<dim3(UPN_ / 128, M_ / 128), 256, 0, stream>>>(
      xz_bf, upw_bf, u_buf, nullptr, nullptr, nullptr, M_, UPN_, XZK_, 0);

  rope_pack_kernel<<<M_, 256, 0, stream>>>(u_buf, next_noise, iaf_up_w, Qb, Kb, Vt);
  fill_reg_pad_kernel<<<B_ * NH_, 256, 0, stream>>>(reg_k, reg_v, Kb, Vt);

  mlp_kernel<<<M_, 256, 0, stream>>>(u_buf, dn_bf);

  attn_kernel<<<dim3(B_ * NH_, S_ / 128), 256, 0, stream>>>(Qb, Kb, Vt, attn_mask, dn_bf);

  gemm_bf16_kernel<<<dim3(D_ / 128, M_ / 128), 256, 0, stream>>>(
      dn_bf, dww_bf, nullptr, h, alpha, out_h, M_, D_, DWK_, 1);
}
